// UMPD_45440753992228
// MI455X (gfx1250) — matrix operand order repaired and checked by reading, not compiled, not run on hardware
//
#include <hip/hip_runtime.h>
#include <hip/hip_bf16.h>

typedef __attribute__((ext_vector_type(16))) _Float16 v16h;
typedef __attribute__((ext_vector_type(8)))  float    v8f;

#define D_MODEL 128
#define C_IN    512
#define NVIEW   7
#define HH      64
#define WW      112
#define HW      (HH*WW)          // 7168
#define PVOX    36864            // 4*96*96
#define EPSBN   1e-5f

// A-matrix element mapping: element i of a lane's v16h holds
// K = half*16 + i  (half = lane>=16), row M = lane%16.
__device__ __forceinline__ int kmap(int i, int half) {
    return (half << 4) + i;
}

// ---------------------------------------------------------------------------
// Prep kernels: pack fp32 weights into WMMA A-operand f16 register layout so
// the GEMM inner loops do one 32B v16h load per WMMA instead of 16 scalar
// loads + converts. All packs are tiny and L2-resident.
// ---------------------------------------------------------------------------
__global__ void p0_pack_wproj(const float* __restrict__ w, _Float16* __restrict__ out)
{
    int t = blockIdx.x * blockDim.x + threadIdx.x;     // 4096 = 8 rowt * 16 kc * 32 lane
    int lane = t & 31, t2 = t >> 5;
    int kc = t2 & 15, rowt = t2 >> 4;
    int half = lane >> 4, lm = lane & 15;
    v16h a;
#pragma unroll
    for (int i = 0; i < 16; ++i)
        a[i] = (_Float16)w[(rowt * 16 + lm) * C_IN + kc * 32 + kmap(i, half)];
    *(v16h*)(out + (size_t)t * 16) = a;
}

__global__ void p1_pack_w3a(const float* __restrict__ w, _Float16* __restrict__ out)
{
    int t = blockIdx.x * blockDim.x + threadIdx.x;     // 1024 = 8 rowt * 4 kc * 32 lane
    int lane = t & 31, t2 = t >> 5;
    int kc = t2 & 3, rowt = t2 >> 2;
    int half = lane >> 4, lm = lane & 15;
    v16h a;
#pragma unroll
    for (int i = 0; i < 16; ++i)
        a[i] = (_Float16)w[(rowt * 16 + lm) * D_MODEL + kc * 32 + kmap(i, half)];
    *(v16h*)(out + (size_t)t * 16) = a;
}

__global__ void p2_pack_w3b(const float* __restrict__ w, _Float16* __restrict__ out)
{
    int t = blockIdx.x * blockDim.x + threadIdx.x;     // 64000 = 4 ot * 125 tap * 4 kc * 32
    int lane = t & 31, t2 = t >> 5;
    int kc = t2 & 3, t3 = t2 >> 2;
    int tap = t3 % 125, ot = t3 / 125;
    int half = lane >> 4, lm = lane & 15;
    v16h a;
#pragma unroll
    for (int i = 0; i < 16; ++i) {
        int c = kc * 32 + kmap(i, half);
        a[i] = (_Float16)w[((size_t)(ot * 16 + lm) * D_MODEL + c) * 125 + tap];
    }
    *(v16h*)(out + ((size_t)((ot * 125 + tap) * 4 + kc) * 32 + lane) * 16) = a;
}

// ---------------------------------------------------------------------------
// Kernel 1: per-view 1x1 conv (128x512 @ 512x7168) + BN1 + ReLU -> feat f16.
// Block = 8 waves = all 8 row-tiles of one (view, pixel-tile); B chunk staged
// once in LDS (double-buffered) and shared by all 8 waves via ds_load.
// grid = 7*448 = 3136 blocks x 256.
// ---------------------------------------------------------------------------
__global__ void k1_proj(const float* __restrict__ img,
                        const _Float16* __restrict__ wprojp,
                        const float* __restrict__ bproj,
                        const float* __restrict__ g,  const float* __restrict__ bb,
                        const float* __restrict__ mm, const float* __restrict__ vv,
                        _Float16* __restrict__ feat)
{
    __shared__ __align__(32) _Float16 bsm[2][16 * 32];   // [buf][n*32 + k], 1KB each

    int tid  = threadIdx.x;
    int wave = tid >> 5, lane = tid & 31;
    int ptile = blockIdx.x % 448;
    int view  = blockIdx.x / 448;
    int rowt = wave;
    int m0 = rowt * 16, p0 = ptile * 16;
    int half = lane >> 4, lm = lane & 15;

    const float* imgv = img + (size_t)view * C_IN * HW;

    // cooperative loader mapping: thread t loads channels 2*lk, 2*lk+1 of pixel ln
    int ln = tid & 15, lk = tid >> 4;

    // prologue: chunk 0
    {
        float f0 = imgv[(size_t)(2 * lk)     * HW + p0 + ln];
        float f1 = imgv[(size_t)(2 * lk + 1) * HW + p0 + ln];
        bsm[0][ln * 32 + 2 * lk]     = (_Float16)f0;
        bsm[0][ln * 32 + 2 * lk + 1] = (_Float16)f1;
    }
    __syncthreads();

    v8f acc = {};
    for (int kc = 0; kc < 16; ++kc) {
        int cur = kc & 1;
        if (kc < 15) {
            int kb = (kc + 1) * 32;
            if (kc < 14)
                __builtin_prefetch(imgv + (size_t)(kb + 32 + 2 * lk) * HW + p0 + ln, 0, 0);
            float f0 = imgv[(size_t)(kb + 2 * lk)     * HW + p0 + ln];
            float f1 = imgv[(size_t)(kb + 2 * lk + 1) * HW + p0 + ln];
            bsm[cur ^ 1][ln * 32 + 2 * lk]     = (_Float16)f0;
            bsm[cur ^ 1][ln * 32 + 2 * lk + 1] = (_Float16)f1;
        }
        v16h a = *(const v16h*)(wprojp + ((size_t)(rowt * 16 + kc) * 32 + lane) * 16);
        v16h b = *(const v16h*)(&bsm[cur][lm * 32 + (half << 4)]);
        acc = __builtin_amdgcn_wmma_f32_16x16x32_f16(false, a, false, b,
                                                     (short)0, acc, false, false);
        __syncthreads();
    }

    int p = p0 + lm;
#pragma unroll
    for (int r = 0; r < 8; ++r) {
        int d = m0 + half * 8 + r;
        float x = acc[r] + bproj[d];
        x = (x - mm[d]) * (g[d] * rsqrtf(vv[d] + EPSBN)) + bb[d];
        x = fmaxf(x, 0.f);
        feat[((size_t)view * D_MODEL + d) * HW + p] = (_Float16)x;
    }
}

// ---------------------------------------------------------------------------
// Kernel 2: backproject gather + softmax-weighted fusion over 7 views.
// fusedt stored voxel-major [p][c] f16. 18432 blocks x 256.
// ---------------------------------------------------------------------------
__global__ void k2_fuse(const _Float16* __restrict__ feat,
                        const int* __restrict__ xidx,
                        const int* __restrict__ yidx,
                        const unsigned char* __restrict__ valid,
                        _Float16* __restrict__ fusedt)
{
    int tid = blockIdx.x * blockDim.x + threadIdx.x;
    int d = tid & (D_MODEL - 1);
    int p = tid >> 7;

    float v[NVIEW];
#pragma unroll
    for (int n = 0; n < NVIEW; ++n) {
        int lin = yidx[n * PVOX + p] * WW + xidx[n * PVOX + p];
        float gv = (float)feat[((size_t)n * D_MODEL + d) * HW + lin];
        v[n] = valid[n * PVOX + p] ? gv : 0.f;
    }
    float mx = v[0];
#pragma unroll
    for (int n = 1; n < NVIEW; ++n) mx = fmaxf(mx, v[n]);
    float den = 0.f, num = 0.f;
#pragma unroll
    for (int n = 0; n < NVIEW; ++n) {
        float e = __expf(v[n] - mx);
        den += e;
        num += e * v[n];
    }
    fusedt[(size_t)p * D_MODEL + d] = (_Float16)(num / den);
}

// ---------------------------------------------------------------------------
// Kernel 3: 1x1x1 conv (128x128 @ 128x36864) + BN2 + ReLU -> h1t [p][c] f16.
// A prepacked; B one v16h load. 2304 blocks x 256 (8 waves = 8 pixel-tiles of one row-tile).
// ---------------------------------------------------------------------------
__global__ void k3_gemm(const _Float16* __restrict__ fusedt,
                        const _Float16* __restrict__ w3ap,
                        const float* __restrict__ b3a,
                        const float* __restrict__ g,  const float* __restrict__ bb,
                        const float* __restrict__ mm, const float* __restrict__ vv,
                        _Float16* __restrict__ h1t)
{
    int wave  = blockIdx.x * (blockDim.x >> 5) + (threadIdx.x >> 5);
    int lane  = threadIdx.x & 31;
    int ptile = wave % 2304;
    int rowt  = wave / 2304;
    int m0 = rowt * 16, p0 = ptile * 16;
    int half = lane >> 4, lm = lane & 15;

    v8f acc = {};
#pragma unroll
    for (int kc = 0; kc < 4; ++kc) {
        v16h a = *(const v16h*)(w3ap + ((size_t)(rowt * 4 + kc) * 32 + lane) * 16);
        v16h b = *(const v16h*)(fusedt + (size_t)(p0 + lm) * D_MODEL + kc * 32 + (half << 4));
        acc = __builtin_amdgcn_wmma_f32_16x16x32_f16(false, a, false, b,
                                                     (short)0, acc, false, false);
    }

    int p = p0 + lm;
#pragma unroll
    for (int r = 0; r < 8; ++r) {
        int d = m0 + half * 8 + r;
        float x = acc[r] + b3a[d];
        x = (x - mm[d]) * (g[d] * rsqrtf(vv[d] + EPSBN)) + bb[d];
        x = fmaxf(x, 0.f);
        h1t[(size_t)p * D_MODEL + d] = (_Float16)x;
    }
}

// ---------------------------------------------------------------------------
// Kernel 4: 5x5x5 conv 128->64 as implicit GEMM. Compute hotspot (75.5 GFLOP).
// Block = 4 waves = 4 output-channel tiles of one 16-voxel z-tile; the 20-voxel
// x 128-ch halo z-row (5KB f16) is staged in LDS once per (kd,kh) via
// CDNA5 async global->LDS copies, zero-padded at borders; inner loop is then
// branch-free ds_load + wmma (EXEC all-1s at every WMMA).
// grid = 2304 blocks x 128.
// ---------------------------------------------------------------------------
__global__ void k4_conv(const _Float16* __restrict__ h1t,
                        const _Float16* __restrict__ w3bp,
                        const float* __restrict__ b3b,
                        const float* __restrict__ g,  const float* __restrict__ bb,
                        const float* __restrict__ mm, const float* __restrict__ vv,
                        float* __restrict__ h2t)
{
    __shared__ __align__(32) _Float16 rowbuf[20 * D_MODEL];   // 5120 B

    int tid  = threadIdx.x;                // 128 threads
    int wave = tid >> 5, lane = tid & 31;
    int ot = wave;
    int m0 = ot * 16;
    int ptile = blockIdx.x;
    int p0 = ptile * 16;
    int half = lane >> 4, lm = lane & 15;

    // whole z-tile shares (x,y); z = z0 + lm
    int x = p0 / 9216;
    int rem0 = p0 - x * 9216;
    int y = rem0 / 96;
    int z0 = rem0 - y * 96;
    int p = p0 + lm;

    v8f acc = {};

    for (int kd = 0; kd < 5; ++kd) {
        int ix = x + kd - 2;
        bool okx = (unsigned)ix < 4u;
        for (int kh = 0; kh < 5; ++kh) {
            int iy = y + kh - 2;
            bool rowok = okx && ((unsigned)iy < 96u);   // block-uniform
            if (rowok) {
                // stage 20 voxels x 128 ch (f16) = 320 x 16B chunks into LDS
                const size_t rowbase = ((size_t)(ix * 96 + iy)) * 96;
                for (int e = tid; e < 320; e += 128) {
                    int vx  = e >> 4;            // voxel 0..19 (iz = z0-2+vx)
                    int o8  = (e & 15) << 3;     // half offset within channel dim
                    int iz  = z0 - 2 + vx;
                    _Float16* dstp = &rowbuf[vx * D_MODEL + o8];
                    if ((unsigned)iz < 96u) {
                        unsigned ldsa = (unsigned)(size_t)dstp;
                        const _Float16* src = h1t + (rowbase + iz) * D_MODEL + o8;
                        asm volatile("global_load_async_to_lds_b128 %0, %1, off"
                                     :: "v"(ldsa), "v"(src) : "memory");
                    } else {
                        uint4 zz = {0u, 0u, 0u, 0u};
                        *(uint4*)dstp = zz;      // zero-pad z border
                    }
                }
                asm volatile("s_wait_asynccnt 0x0" ::: "memory");
                __syncthreads();

                // 5 taps along z * 4 K-chunks = 20 WMMAs, all fed from LDS
#pragma unroll
                for (int kw = 0; kw < 5; ++kw) {
                    int tap = (kd * 5 + kh) * 5 + kw;
#pragma unroll
                    for (int kc = 0; kc < 4; ++kc) {
                        v16h a = *(const v16h*)(w3bp +
                                  ((size_t)((ot * 125 + tap) * 4 + kc) * 32 + lane) * 16);
                        v16h b = *(const v16h*)(&rowbuf[(lm + kw) * D_MODEL +
                                                        kc * 32 + (half << 4)]);
                        acc = __builtin_amdgcn_wmma_f32_16x16x32_f16(
                                  false, a, false, b, (short)0, acc, false, false);
                    }
                }
                __syncthreads();
            }
        }
    }

#pragma unroll
    for (int r = 0; r < 8; ++r) {
        int o = m0 + half * 8 + r;
        float v = acc[r] + b3b[o];
        v = (v - mm[o]) * (g[o] * rsqrtf(vv[o] + EPSBN)) + bb[o];
        v = fmaxf(v, 0.f);
        h2t[(size_t)p * 64 + o] = v;
    }
}

// ---------------------------------------------------------------------------
// Kernel 5: final 1x1x1 conv 64->4 + sigmoid -> (4, P) channel-major output
// (== split([3,1]) concatenated flat). 576 blocks x 256.
// ---------------------------------------------------------------------------
__global__ void k5_head(const float* __restrict__ h2t,
                        const float* __restrict__ w3c,
                        const float* __restrict__ b3c,
                        float* __restrict__ out)
{
    int tid = blockIdx.x * blockDim.x + threadIdx.x;
    int o = tid & 3;
    int p = tid >> 2;
    float s = b3c[o];
#pragma unroll
    for (int c = 0; c < 64; ++c)
        s += w3c[o * 64 + c] * h2t[(size_t)p * 64 + c];
    out[(size_t)o * PVOX + p] = 1.f / (1.f + __expf(-s));
}

// ---------------------------------------------------------------------------
extern "C" void kernel_launch(void* const* d_in, const int* in_sizes, int n_in,
                              void* d_out, int out_size, void* d_ws, size_t ws_size,
                              hipStream_t stream)
{
    const float*         img   = (const float*)d_in[0];
    const int*           xidx  = (const int*)d_in[1];
    const int*           yidx  = (const int*)d_in[2];
    const unsigned char* valid = (const unsigned char*)d_in[3];
    const float* wproj = (const float*)d_in[4];
    const float* bproj = (const float*)d_in[5];
    const float* bn1g  = (const float*)d_in[6];
    const float* bn1b  = (const float*)d_in[7];
    const float* bn1m  = (const float*)d_in[8];
    const float* bn1v  = (const float*)d_in[9];
    const float* w3a   = (const float*)d_in[10];
    const float* b3a   = (const float*)d_in[11];
    const float* bn2g  = (const float*)d_in[12];
    const float* bn2b  = (const float*)d_in[13];
    const float* bn2m  = (const float*)d_in[14];
    const float* bn2v  = (const float*)d_in[15];
    const float* w3b   = (const float*)d_in[16];
    const float* b3b   = (const float*)d_in[17];
    const float* bn3g  = (const float*)d_in[18];
    const float* bn3b  = (const float*)d_in[19];
    const float* bn3m  = (const float*)d_in[20];
    const float* bn3v  = (const float*)d_in[21];
    const float* w3c   = (const float*)d_in[22];
    const float* b3c   = (const float*)d_in[23];

    char* ws = (char*)d_ws;
    const size_t FEAT_B   = (size_t)NVIEW * D_MODEL * HW * 2;  // 12,845,056
    const size_t FUSED_B  = (size_t)PVOX * D_MODEL * 2;        //  9,437,184
    const size_t H1_B     = FUSED_B;
    const size_t H2_B     = (size_t)PVOX * 64 * 4;             //  9,437,184
    const size_t WPROJP_B = 8 * 16 * 32 * 16 * 2;              //    131,072
    const size_t W3AP_B   = 8 * 4 * 32 * 16 * 2;               //     32,768

    _Float16* feat   = (_Float16*)ws;
    _Float16* fusedt = (_Float16*)(ws + FEAT_B);
    _Float16* h1t    = (_Float16*)(ws + FEAT_B + FUSED_B);
    float*    h2t    = (float*)   (ws + FEAT_B + FUSED_B + H1_B);
    _Float16* wprojp = (_Float16*)(ws + FEAT_B + FUSED_B + H1_B + H2_B);
    _Float16* w3ap   = (_Float16*)(ws + FEAT_B + FUSED_B + H1_B + H2_B + WPROJP_B);
    _Float16* w3bp   = (_Float16*)(ws + FEAT_B + FUSED_B + H1_B + H2_B + WPROJP_B + W3AP_B);

    float* out = (float*)d_out;

    p0_pack_wproj<<<16, 256, 0, stream>>>(wproj, wprojp);
    p1_pack_w3a  <<<4, 256, 0, stream>>>(w3a, w3ap);
    p2_pack_w3b  <<<250, 256, 0, stream>>>(w3b, w3bp);

    k1_proj<<<3136, 256, 0, stream>>>(img, wprojp, bproj, bn1g, bn1b, bn1m, bn1v, feat);
    k2_fuse<<<18432, 256, 0, stream>>>(feat, xidx, yidx, valid, fusedt);
    k3_gemm<<<2304, 256, 0, stream>>>(fusedt, w3ap, b3a, bn2g, bn2b, bn2m, bn2v, h1t);
    k4_conv<<<2304, 128, 0, stream>>>(h1t, w3bp, b3b, bn3g, bn3b, bn3m, bn3v, h2t);
    k5_head<<<576, 256, 0, stream>>>(h2t, w3c, b3c, out);
}
